// PointNet_26594437497543
// MI455X (gfx1250) — matrix operand order repaired and checked by reading, not compiled, not run on hardware
//
#include <hip/hip_runtime.h>
#include <hip/hip_bf16.h>

typedef _Float16 h8_t  __attribute__((ext_vector_type(8)));
typedef _Float16 h16_t __attribute__((ext_vector_type(16)));
typedef float    f8_t  __attribute__((ext_vector_type(8)));

#define BATCH 8
#define NPTS  2048
#define KNBR  64
#define S1N   1024
#define S2N   256

__device__ __constant__ const float kBnInv = 0.9999950000374997f; // 1/sqrt(1+1e-5)

union AB16 { h16_t v; h8_t h[2]; };

struct LayerP { const _Float16* w; const float* b; const float* g; const float* be; };

// ---------------------------------------------------------------------------
// Weight repack: W[cin][cout] f32 -> wT[cout][cinPad] f16 (zero padded K)
// ---------------------------------------------------------------------------
__global__ __launch_bounds__(256) void prep_wT(const float* __restrict__ W,
    int cin, int cout, int cinP, _Float16* __restrict__ out) {
  int e = blockIdx.x * blockDim.x + threadIdx.x;
  int tot = cout * cinP;
  if (e >= tot) return;
  int co = e / cinP, ci = e % cinP;
  float v = (ci < cin) ? W[(size_t)ci * cout + co] : 0.0f;
  out[e] = (_Float16)v;
}

// ---------------------------------------------------------------------------
// Farthest point sampling: one block per event. Sequential argmax steps.
// ---------------------------------------------------------------------------
__global__ __launch_bounds__(256) void fps_kernel(const float* __restrict__ pos,
    int Pin, int nS, int* __restrict__ idx, float* __restrict__ qpos) {
  __shared__ float px[NPTS], py[NPTS], pz[NPTS], dmin[NPTS];
  __shared__ float rv[256];
  __shared__ int   ri[256];
  int b = blockIdx.x, tid = threadIdx.x;
  const float* pb = pos + (size_t)b * Pin * 3;
  for (int j = tid; j < Pin; j += 256) {
    px[j] = pb[j * 3 + 0]; py[j] = pb[j * 3 + 1]; pz[j] = pb[j * 3 + 2];
  }
  __syncthreads();
  float x0 = px[0], y0 = py[0], z0 = pz[0];
  for (int j = tid; j < Pin; j += 256) {
    float dx = px[j] - x0, dy = py[j] - y0, dz = pz[j] - z0;
    dmin[j] = dx * dx + dy * dy + dz * dz;
  }
  if (tid == 0) {
    idx[(size_t)b * nS] = 0;
    qpos[(size_t)b * nS * 3 + 0] = x0;
    qpos[(size_t)b * nS * 3 + 1] = y0;
    qpos[(size_t)b * nS * 3 + 2] = z0;
  }
  __syncthreads();
  for (int i = 1; i < nS; ++i) {
    float bv = -1.0f; int bi = 0;
    for (int j = tid; j < Pin; j += 256) {
      float d = dmin[j];
      if (d > bv || (d == bv && j < bi)) { bv = d; bi = j; }
    }
    rv[tid] = bv; ri[tid] = bi;
    __syncthreads();
    for (int s = 128; s > 0; s >>= 1) {
      if (tid < s) {
        float ov = rv[tid + s]; int oi = ri[tid + s];
        if (ov > rv[tid] || (ov == rv[tid] && oi < ri[tid])) { rv[tid] = ov; ri[tid] = oi; }
      }
      __syncthreads();
    }
    int sel = ri[0];
    float sx = px[sel], sy = py[sel], sz = pz[sel];
    for (int j = tid; j < Pin; j += 256) {
      float dx = px[j] - sx, dy = py[j] - sy, dz = pz[j] - sz;
      dmin[j] = fminf(dmin[j], dx * dx + dy * dy + dz * dz);
    }
    if (tid == 0) {
      idx[(size_t)b * nS + i] = sel;
      qpos[((size_t)b * nS + i) * 3 + 0] = sx;
      qpos[((size_t)b * nS + i) * 3 + 1] = sy;
      qpos[((size_t)b * nS + i) * 3 + 2] = sz;
    }
    __syncthreads();
  }
}

// ---------------------------------------------------------------------------
// Ball query: one block per (b,s). K rounds of argmin (= K smallest in radius)
// ---------------------------------------------------------------------------
__global__ __launch_bounds__(256) void ball_query_kernel(const float* __restrict__ qpos,
    const float* __restrict__ pos, int S, int Pin, float r2,
    int* __restrict__ nidx, unsigned char* __restrict__ valid) {
  __shared__ float d2[NPTS];
  __shared__ float rv[256];
  __shared__ int   ri[256];
  int bs = blockIdx.x, tid = threadIdx.x;
  int b = bs / S;
  const float FINF = __builtin_inff();
  const float* pb = pos + (size_t)b * Pin * 3;
  float qx = qpos[(size_t)bs * 3 + 0];
  float qy = qpos[(size_t)bs * 3 + 1];
  float qz = qpos[(size_t)bs * 3 + 2];
  for (int j = tid; j < Pin; j += 256) {
    float dx = pb[j * 3 + 0] - qx, dy = pb[j * 3 + 1] - qy, dz = pb[j * 3 + 2] - qz;
    float d = dx * dx + dy * dy + dz * dz;
    d2[j] = (d <= r2) ? d : FINF;
  }
  __syncthreads();
  for (int t = 0; t < KNBR; ++t) {
    float bv = FINF; int bi = Pin;
    for (int j = tid; j < Pin; j += 256) {
      float d = d2[j];
      if (d < bv || (d == bv && j < bi)) { bv = d; bi = j; }
    }
    rv[tid] = bv; ri[tid] = bi;
    __syncthreads();
    for (int s = 128; s > 0; s >>= 1) {
      if (tid < s) {
        float ov = rv[tid + s]; int oi = ri[tid + s];
        if (ov < rv[tid] || (ov == rv[tid] && oi < ri[tid])) { rv[tid] = ov; ri[tid] = oi; }
      }
      __syncthreads();
    }
    if (tid == 0) {
      float v = rv[0]; int s0 = ri[0];
      int ok = (v < FINF) ? 1 : 0;
      nidx[(size_t)bs * KNBR + t]  = ok ? s0 : 0;
      valid[(size_t)bs * KNBR + t] = (unsigned char)ok;
      if (ok) d2[s0] = FINF;
    }
    __syncthreads();
  }
}

// ---------------------------------------------------------------------------
// WMMA 16x16 output tile: A from LDS (16 rows x cinP f16, row-major),
// B from global prepped weights wT[cout][cinP]. Accumulates over cinP in K=32.
// A layout: lane M = lane&15; K-halves {kk+hi*8 .. +7} and {kk+16+hi*8 .. +7}
// B layout: lane N = lane&15; K-halves {kk+hi*8 .. +7} and {kk+16+hi*8 .. +7}
// D layout: lane col = lane&15; reg r = row r + 8*hi
// ---------------------------------------------------------------------------
__device__ __forceinline__ f8_t wmma_tile(const _Float16* src, int cinP,
    const _Float16* wT, float bv, int m0, int lane, int oc) {
  int col = lane & 15, hi = lane >> 4;
  f8_t acc;
#pragma unroll
  for (int i = 0; i < 8; ++i) acc[i] = bv;
  const _Float16* srow = src + (size_t)(m0 + col) * cinP;
  const _Float16* wrow = wT + (size_t)oc * cinP;
  for (int kk = 0; kk < cinP; kk += 32) {
    AB16 A, B;
    A.h[0] = *(const h8_t*)(srow + kk + hi * 8);
    A.h[1] = *(const h8_t*)(srow + kk + hi * 8 + 16);
    B.h[0] = *(const h8_t*)(wrow + kk + hi * 8);
    B.h[1] = *(const h8_t*)(wrow + kk + hi * 8 + 16);
    acc = __builtin_amdgcn_wmma_f32_16x16x32_f16(false, A.v, false, B.v,
                                                 (short)0, acc, false, false);
  }
  return acc;
}

// One MLP layer: y = BN(relu(xW+b)) -> f16 LDS tile [.. rows][cout]
__device__ __forceinline__ void mlp_layer(const _Float16* src, int cinP,
    LayerP L, int cout, _Float16* dst, int m0, int lane) {
  int col = lane & 15, hi = lane >> 4;
  for (int n0 = 0; n0 < cout; n0 += 16) {
    int oc = n0 + col;
    f8_t acc = wmma_tile(src, cinP, L.w, L.b[oc], m0, lane, oc);
    float gs = L.g[oc] * kBnInv, bb = L.be[oc];
#pragma unroll
    for (int r = 0; r < 8; ++r) {
      float v = acc[r];
      v = v > 0.0f ? v : 0.0f;
      v = v * gs + bb;
      dst[(size_t)(m0 + r + hi * 8) * cout + oc] = (_Float16)v;
    }
  }
}

// Final MLP layer reduced to per-column maxes straight from D fragments.
__device__ __forceinline__ void mlp_layer_max(const _Float16* src, int cinP,
    LayerP L, int cout, const unsigned char* vld, float* maxbuf,
    int m0, int wave, int lane) {
  int col = lane & 15, hi = lane >> 4;
  const float FINF = __builtin_inff();
  for (int n0 = 0; n0 < cout; n0 += 16) {
    int oc = n0 + col;
    f8_t acc = wmma_tile(src, cinP, L.w, L.b[oc], m0, lane, oc);
    float gs = L.g[oc] * kBnInv, bb = L.be[oc];
    float m = -FINF;
#pragma unroll
    for (int r = 0; r < 8; ++r) {
      float v = acc[r];
      v = v > 0.0f ? v : 0.0f;
      v = v * gs + bb;
      int k = m0 + r + hi * 8;
      if (vld != nullptr && !vld[k]) v = -FINF;
      m = fmaxf(m, v);
    }
    float o = __shfl_xor(m, 16, 32);  // combine the two 16-lane halves (rows r vs r+8)
    m = fmaxf(m, o);
    if ((lane & 16) == 0) maxbuf[(size_t)wave * cout + oc] = m;
  }
}

// ---------------------------------------------------------------------------
// SA module conv: one block (4 waves) per centroid; 64 neighbor rows.
// feat = [x_j (CX), pos_j - pos_i (3), zero-pad to CINP]
// ---------------------------------------------------------------------------
template<int CX, int CINP, int C1, int C2, int C3>
__global__ __launch_bounds__(128) void sa_conv_kernel(
    const float* __restrict__ pos, const _Float16* __restrict__ xin, int Pin, int S,
    const float* __restrict__ qpos, const int* __restrict__ nidx,
    const unsigned char* __restrict__ valid,
    LayerP L1, LayerP L2, LayerP L3,
    _Float16* __restrict__ xout) {
  __shared__ _Float16 bufIn[64 * CINP];
  __shared__ _Float16 bufA[64 * C1];
  __shared__ _Float16 bufB[64 * C2];
  __shared__ float maxbuf[4 * C3];
  __shared__ int nb[64];
  __shared__ unsigned char vld[64];
  int bs = blockIdx.x, tid = threadIdx.x;
  int b = bs / S;
  int lane = tid & 31, wave = tid >> 5;
  if (tid < 64) {
    nb[tid]  = nidx[(size_t)bs * KNBR + tid];
    vld[tid] = valid[(size_t)bs * KNBR + tid];
  }
  __syncthreads();
  float q0 = qpos[(size_t)bs * 3 + 0];
  float q1 = qpos[(size_t)bs * 3 + 1];
  float q2 = qpos[(size_t)bs * 3 + 2];
  for (int e = tid; e < 64 * CINP; e += 128) {
    int k = e / CINP, c = e % CINP;
    int pj = nb[k];
    float v = 0.0f;
    if (c < CX) {
      v = (float)xin[((size_t)b * Pin + pj) * CX + c];
    } else if (c < CX + 3) {
      int a = c - CX;
      float pv = pos[((size_t)b * Pin + pj) * 3 + a];
      float qv = (a == 0) ? q0 : ((a == 1) ? q1 : q2);
      v = pv - qv;
    }
    bufIn[e] = (_Float16)v;
  }
  __syncthreads();
  int m0 = wave * 16;
  mlp_layer(bufIn, CINP, L1, C1, bufA, m0, lane);
  __syncthreads();
  mlp_layer(bufA, C1, L2, C2, bufB, m0, lane);
  __syncthreads();
  mlp_layer_max(bufB, C2, L3, C3, vld, maxbuf, m0, wave, lane);
  __syncthreads();
  for (int c = tid; c < C3; c += 128) {
    float m = fmaxf(fmaxf(maxbuf[c], maxbuf[C3 + c]),
                    fmaxf(maxbuf[2 * C3 + c], maxbuf[3 * C3 + c]));
    xout[(size_t)bs * C3 + c] = (_Float16)m;
  }
}

// ---------------------------------------------------------------------------
// SA3 global MLP: feat = [x2 (256), p2 (3)] -> 256 -> 512 -> 1024, partial max
// One block = 2 waves = 32 rows; 8 row-tiles per event.
// ---------------------------------------------------------------------------
__global__ __launch_bounds__(64) void sa3_kernel(
    const _Float16* __restrict__ x2, const float* __restrict__ q2,
    LayerP L1, LayerP L2, LayerP L3,
    float* __restrict__ part) {
  __shared__ _Float16 regA[32 * 512]; // input (32x288) then h2 (32x512)
  __shared__ _Float16 regB[32 * 256]; // h1
  __shared__ float maxbuf[2 * 1024];
  int blk = blockIdx.x, tid = threadIdx.x;
  int b = blk / 8, tile = blk % 8;
  int lane = tid & 31, wave = tid >> 5;
  for (int e = tid; e < 32 * 288; e += 64) {
    int rrow = e / 288, c = e % 288;
    int p = tile * 32 + rrow;
    float v = 0.0f;
    if (c < 256)      v = (float)x2[((size_t)b * S2N + p) * 256 + c];
    else if (c < 259) v = q2[((size_t)b * S2N + p) * 3 + (c - 256)];
    regA[(size_t)rrow * 288 + c] = (_Float16)v;
  }
  __syncthreads();
  int m0 = wave * 16;
  mlp_layer(regA, 288, L1, 256, regB, m0, lane);
  __syncthreads();
  mlp_layer(regB, 256, L2, 512, regA, m0, lane);
  __syncthreads();
  mlp_layer_max(regA, 512, L3, 1024, nullptr, maxbuf, m0, wave, lane);
  __syncthreads();
  for (int c = tid; c < 1024; c += 64) {
    part[(size_t)blk * 1024 + c] = fmaxf(maxbuf[c], maxbuf[1024 + c]);
  }
}

__global__ __launch_bounds__(256) void greduce_kernel(const float* __restrict__ part,
                                                      float* __restrict__ g) {
  int b = blockIdx.x, tid = threadIdx.x;
  for (int c = tid; c < 1024; c += 256) {
    float m = -__builtin_inff();
    for (int t = 0; t < 8; ++t) m = fmaxf(m, part[((size_t)b * 8 + t) * 1024 + c]);
    g[(size_t)b * 1024 + c] = m;
  }
}

// ---------------------------------------------------------------------------
// Classifier head: 1024 -> 512 -> 256 -> 10, one block per event.
// ---------------------------------------------------------------------------
__global__ __launch_bounds__(256) void head_kernel(const float* __restrict__ g,
    const float* __restrict__ W1, const float* __restrict__ b1,
    const float* __restrict__ W2, const float* __restrict__ b2,
    const float* __restrict__ W3, const float* __restrict__ b3,
    float* __restrict__ out) {
  __shared__ float gl[1024];
  __shared__ float h1[512];
  __shared__ float h2[256];
  int b = blockIdx.x, tid = threadIdx.x;
  for (int c = tid; c < 1024; c += 256) gl[c] = g[(size_t)b * 1024 + c];
  __syncthreads();
  for (int o = tid; o < 512; o += 256) {
    float s = b1[o];
    for (int i = 0; i < 1024; ++i) s += gl[i] * W1[(size_t)i * 512 + o];
    h1[o] = s > 0.0f ? s : 0.0f;
  }
  __syncthreads();
  for (int o = tid; o < 256; o += 256) {
    float s = b2[o];
    for (int i = 0; i < 512; ++i) s += h1[i] * W2[(size_t)i * 256 + o];
    h2[o] = s > 0.0f ? s : 0.0f;
  }
  __syncthreads();
  if (tid < 10) {
    float s = b3[tid];
    for (int i = 0; i < 256; ++i) s += h2[i] * W3[(size_t)i * 10 + tid];
    out[b * 10 + tid] = s;
  }
}

// ---------------------------------------------------------------------------
extern "C" void kernel_launch(void* const* d_in, const int* in_sizes, int n_in,
                              void* d_out, int out_size, void* d_ws, size_t ws_size,
                              hipStream_t stream) {
  (void)in_sizes; (void)n_in; (void)out_size; (void)ws_size;
  const float* pos = (const float*)d_in[0];
  // d_in[1] = batch (unused; structure is static)
  auto F = [&](int i) { return (const float*)d_in[i]; };

  // ---- workspace carve ----
  char* ws = (char*)d_ws;
  size_t off = 0;
  auto carve = [&](size_t bytes) -> char* {
    char* p = ws + off;
    off += bytes;
    off = (off + 255) & ~(size_t)255;
    return p;
  };
  int*            idx1   = (int*)           carve((size_t)BATCH * S1N * 4);
  float*          q1     = (float*)         carve((size_t)BATCH * S1N * 3 * 4);
  int*            nidx1  = (int*)           carve((size_t)BATCH * S1N * KNBR * 4);
  unsigned char*  valid1 = (unsigned char*) carve((size_t)BATCH * S1N * KNBR);
  _Float16*       x1     = (_Float16*)      carve((size_t)BATCH * S1N * 128 * 2);
  int*            idx2   = (int*)           carve((size_t)BATCH * S2N * 4);
  float*          q2     = (float*)         carve((size_t)BATCH * S2N * 3 * 4);
  int*            nidx2  = (int*)           carve((size_t)BATCH * S2N * KNBR * 4);
  unsigned char*  valid2 = (unsigned char*) carve((size_t)BATCH * S2N * KNBR);
  _Float16*       x2     = (_Float16*)      carve((size_t)BATCH * S2N * 256 * 2);
  float*          part3  = (float*)         carve((size_t)BATCH * 8 * 1024 * 4);
  float*          gvec   = (float*)         carve((size_t)BATCH * 1024 * 4);
  _Float16* w10 = (_Float16*)carve((size_t)64   * 32  * 2);
  _Float16* w11 = (_Float16*)carve((size_t)64   * 64  * 2);
  _Float16* w12 = (_Float16*)carve((size_t)128  * 64  * 2);
  _Float16* w20 = (_Float16*)carve((size_t)128  * 160 * 2);
  _Float16* w21 = (_Float16*)carve((size_t)128  * 128 * 2);
  _Float16* w22 = (_Float16*)carve((size_t)256  * 128 * 2);
  _Float16* w30 = (_Float16*)carve((size_t)256  * 288 * 2);
  _Float16* w31 = (_Float16*)carve((size_t)512  * 256 * 2);
  _Float16* w32 = (_Float16*)carve((size_t)1024 * 512 * 2);

  // ---- weight repack (f32 -> transposed padded f16) ----
  auto prep = [&](const float* W, int cin, int cout, int cinP, _Float16* o) {
    int tot = cout * cinP;
    prep_wT<<<(tot + 255) / 256, 256, 0, stream>>>(W, cin, cout, cinP, o);
  };
  prep(F(2),  3,   64,   32,  w10);
  prep(F(6),  64,  64,   64,  w11);
  prep(F(10), 64,  128,  64,  w12);
  prep(F(14), 131, 128,  160, w20);
  prep(F(18), 128, 128,  128, w21);
  prep(F(22), 128, 256,  128, w22);
  prep(F(26), 259, 256,  288, w30);
  prep(F(30), 256, 512,  256, w31);
  prep(F(34), 512, 1024, 512, w32);

  LayerP s1L1 = { w10, F(3),  F(4),  F(5)  };
  LayerP s1L2 = { w11, F(7),  F(8),  F(9)  };
  LayerP s1L3 = { w12, F(11), F(12), F(13) };
  LayerP s2L1 = { w20, F(15), F(16), F(17) };
  LayerP s2L2 = { w21, F(19), F(20), F(21) };
  LayerP s2L3 = { w22, F(23), F(24), F(25) };
  LayerP s3L1 = { w30, F(27), F(28), F(29) };
  LayerP s3L2 = { w31, F(31), F(32), F(33) };
  LayerP s3L3 = { w32, F(35), F(36), F(37) };

  // ---- SA1 ----
  fps_kernel<<<BATCH, 256, 0, stream>>>(pos, NPTS, S1N, idx1, q1);
  ball_query_kernel<<<BATCH * S1N, 256, 0, stream>>>(q1, pos, S1N, NPTS,
                                                     0.2f * 0.2f, nidx1, valid1);
  sa_conv_kernel<0, 32, 64, 64, 128><<<BATCH * S1N, 128, 0, stream>>>(
      pos, (const _Float16*)nullptr, NPTS, S1N, q1, nidx1, valid1,
      s1L1, s1L2, s1L3, x1);

  // ---- SA2 ----
  fps_kernel<<<BATCH, 256, 0, stream>>>(q1, S1N, S2N, idx2, q2);
  ball_query_kernel<<<BATCH * S2N, 256, 0, stream>>>(q2, q1, S2N, S1N,
                                                     0.4f * 0.4f, nidx2, valid2);
  sa_conv_kernel<128, 160, 128, 128, 256><<<BATCH * S2N, 128, 0, stream>>>(
      q1, x1, S1N, S2N, q2, nidx2, valid2,
      s2L1, s2L2, s2L3, x2);

  // ---- SA3 global ----
  sa3_kernel<<<BATCH * 8, 64, 0, stream>>>(x2, q2, s3L1, s3L2, s3L3, part3);
  greduce_kernel<<<BATCH, 256, 0, stream>>>(part3, gvec);

  // ---- head ----
  head_kernel<<<BATCH, 256, 0, stream>>>(gvec,
      F(38), F(39), F(40), F(41), F(42), F(43), (float*)d_out);
}
